// DeformableConv2d_21560735826439
// MI455X (gfx1250) — compile-verified, matrix operand order checked by reading, not run on hardware
//
#include <hip/hip_runtime.h>
#include <hip/hip_bf16.h>

// ---------------------------------------------------------------------------
// Deformable conv (B=8, C=64, H=W=128, K=9, Cout=64) for gfx1250 / MI455X.
//
// Stage A (deform_sample_kernel): one wave32 per output pixel.
//   - depthwise 3x3 conv -> 1152 offset logits (weights staged in LDS)
//   - wave-wide softmax over the 1152 logits (shfl_xor tree, wave32)
//   - bilinear sampling with zero padding
//   - emit sampled activations as f16 row-major [M=131072, CK=576] (WMMA A)
// Stage B0 (pack_w_kernel): swizzle deform_w (f32) into the exact WMMA
//   B-fragment lane layout as f16 so the GEMM does contiguous v16h loads.
// Stage B (wmma_gemm_kernel): M=131072, Kdim=576, N=64 GEMM with
//   v_wmma_f32_16x16x32_f16; one 16x16 tile per wave, 18 k-steps of 32.
// ---------------------------------------------------------------------------

#define Bn   8
#define Cn   64
#define Hn   128
#define Wn   128
#define HWn  (Hn * Wn)          // 16384
#define Kn   9
#define COUT 64
#define CK   (Cn * Kn)          // 576
#define OFFC (2 * Kn * Cn)      // 1152
#define Mn   (Bn * HWn)         // 131072
#define KSTEPS (CK / 32)        // 18

typedef __attribute__((ext_vector_type(16))) _Float16 v16h;
typedef __attribute__((ext_vector_type(8)))  float    v8f;

// ---------------------------------------------------------------------------
// Stage A: offset conv + softmax + bilinear sampling.
// Block = 256 threads = 8 waves; one wave handles one (b,h,w) pixel.
// Lane l owns channels {2l, 2l+1}: 36 offset logits, 18 sampled values.
// ---------------------------------------------------------------------------
__global__ __launch_bounds__(256)
void deform_sample_kernel(const float* __restrict__ x,
                          const float* __restrict__ offw,   // [1152,1,3,3]
                          const float* __restrict__ offb,   // [1152]
                          _Float16* __restrict__ Asamp)     // [M, CK] f16
{
    __shared__ float w_lds[OFFC * 9];   // 41472 B
    __shared__ float b_lds[OFFC];       //  4608 B

    for (int i = threadIdx.x; i < OFFC * 9; i += 256) w_lds[i] = offw[i];
    for (int i = threadIdx.x; i < OFFC;     i += 256) b_lds[i] = offb[i];
    __syncthreads();

    const int wave = threadIdx.x >> 5;
    const int lane = threadIdx.x & 31;
    const int pix  = blockIdx.x * 8 + wave;      // 0 .. M-1
    const int b    = pix >> 14;                  // / HWn
    const int hw   = pix & (HWn - 1);
    const int h    = hw >> 7;                    // / Wn
    const int w    = hw & (Wn - 1);

    float offv[36];

    // -------- depthwise conv: 18 logits per channel, 2 channels per lane ----
#pragma unroll
    for (int cc = 0; cc < 2; ++cc) {
        const int c = lane * 2 + cc;
        const float* __restrict__ xc = x + ((size_t)(b * Cn + c)) * HWn;
        float tap[9];
#pragma unroll
        for (int t = 0; t < 9; ++t) {
            const int ty = h - 1 + t / 3;
            const int tx = w - 1 + t % 3;
            const bool ok = (ty >= 0) & (ty < Hn) & (tx >= 0) & (tx < Wn);
            tap[t] = ok ? xc[ty * Wn + tx] : 0.0f;
        }
#pragma unroll
        for (int j = 0; j < 18; ++j) {
            const int o = c * 18 + j;
            float acc = b_lds[o];
#pragma unroll
            for (int t = 0; t < 9; ++t) acc += tap[t] * w_lds[o * 9 + t];
            offv[cc * 18 + j] = acc;
        }
    }

    // -------- softmax across all 1152 logits of this pixel (wave-wide) -----
    float mx = -3.4e38f;
#pragma unroll
    for (int i = 0; i < 36; ++i) mx = fmaxf(mx, offv[i]);
#pragma unroll
    for (int s = 16; s > 0; s >>= 1) mx = fmaxf(mx, __shfl_xor(mx, s, 32));

    float sum = 0.0f;
#pragma unroll
    for (int i = 0; i < 36; ++i) {
        offv[i] = __expf(offv[i] - mx);
        sum += offv[i];
    }
#pragma unroll
    for (int s = 16; s > 0; s >>= 1) sum += __shfl_xor(sum, s, 32);
    const float inv = 1.0f / sum;

    // -------- bilinear sampling, emit f16 A-matrix rows ---------------------
    _Float16* __restrict__ arow = Asamp + (size_t)pix * CK;
#pragma unroll
    for (int cc = 0; cc < 2; ++cc) {
        const int c = lane * 2 + cc;
        const float* __restrict__ xc = x + ((size_t)(b * Cn + c)) * HWn;
#pragma unroll
        for (int k = 0; k < 9; ++k) {
            const float dy = offv[cc * 18 + 2 * k]     * inv;
            const float dx = offv[cc * 18 + 2 * k + 1] * inv;
            const float py = (float)(h - 1 + k / 3) + dy;
            const float px = (float)(w - 1 + k % 3) + dx;

            const float y0f = floorf(py);
            const float x0f = floorf(px);
            const int   y0  = (int)y0f;
            const int   x0  = (int)x0f;
            const float wy1 = py - y0f, wx1 = px - x0f;
            const float wy0 = 1.0f - wy1, wx0 = 1.0f - wx1;

            float g00 = 0.f, g01 = 0.f, g10 = 0.f, g11 = 0.f;
            const bool yok0 = (y0 >= 0)     & (y0 < Hn);
            const bool yok1 = (y0 + 1 >= 0) & (y0 + 1 < Hn);
            const bool xok0 = (x0 >= 0)     & (x0 < Wn);
            const bool xok1 = (x0 + 1 >= 0) & (x0 + 1 < Wn);
            if (yok0 & xok0) g00 = xc[y0 * Wn + x0];
            if (yok0 & xok1) g01 = xc[y0 * Wn + x0 + 1];
            if (yok1 & xok0) g10 = xc[(y0 + 1) * Wn + x0];
            if (yok1 & xok1) g11 = xc[(y0 + 1) * Wn + x0 + 1];

            const float s = wy0 * (wx0 * g00 + wx1 * g01) +
                            wy1 * (wx0 * g10 + wx1 * g11);
            arow[c * 9 + k] = (_Float16)s;
        }
    }
}

// ---------------------------------------------------------------------------
// Stage B0: pack deform_w [Cout, C*K] f32 into WMMA B-fragment layout, f16.
// Bpack index = (((ntile*18 + kstep)*32 + lane)*16 + j), element is
//   deform_w[n = ntile*16 + (lane&15)][kk = kstep*32 + (lane>>4)*16 + j].
// ---------------------------------------------------------------------------
__global__ __launch_bounds__(256)
void pack_w_kernel(const float* __restrict__ dw, _Float16* __restrict__ Bpack)
{
    const int idx = blockIdx.x * 256 + threadIdx.x;   // 4*18*32*16 = 36864
    if (idx >= 4 * KSTEPS * 32 * 16) return;
    const int j     = idx & 15;
    const int lane  = (idx >> 4) & 31;
    const int kstep = (idx >> 9) % KSTEPS;
    const int ntile = idx / (KSTEPS * 512);
    const int n  = ntile * 16 + (lane & 15);
    const int kk = kstep * 32 + ((lane >> 4) * 16) + j;
    Bpack[idx] = (_Float16)dw[n * CK + kk];
}

// ---------------------------------------------------------------------------
// Stage B: GEMM via v_wmma_f32_16x16x32_f16.
// Block = 256 threads = 8 waves. Wave -> one 16x16 (Mtile, Ntile) output tile:
//   waves 0-3: m-tile 2*blockIdx.x,   n-tiles 0..3
//   waves 4-7: m-tile 2*blockIdx.x+1, n-tiles 0..3
// ---------------------------------------------------------------------------
__global__ __launch_bounds__(256)
void wmma_gemm_kernel(const _Float16* __restrict__ Asamp,
                      const _Float16* __restrict__ Bpack,
                      const float* __restrict__ db,
                      float* __restrict__ out)
{
    const int wave = threadIdx.x >> 5;
    const int lane = threadIdx.x & 31;
    const int mt   = blockIdx.x * 2 + (wave >> 2);    // 0 .. 8191
    const int nt   = wave & 3;                        // 0 .. 3
    const int m0   = mt * 16;

    // A fragment address: lane&15 -> row, lane>>4 -> K half (contiguous k order, same as Bpack)
    const int arow = m0 + (lane & 15);
    const int koff = (lane >> 4) * 16;
    const _Float16* __restrict__ Ap = Asamp + (size_t)arow * CK + koff;
    const _Float16* __restrict__ Bp = Bpack + (size_t)nt * (KSTEPS * 512)
                                            + (size_t)lane * 16;

    v8f acc = {};
#pragma unroll
    for (int ks = 0; ks < KSTEPS; ++ks) {
        if (ks + 2 < KSTEPS)
            __builtin_prefetch(Ap + (ks + 2) * 32, 0, 1);  // global_prefetch_b8
        const v16h a = *(const v16h*)(Ap + ks * 32);
        const v16h bf = *(const v16h*)(Bp + ks * 512);
        acc = __builtin_amdgcn_wmma_f32_16x16x32_f16(
            /*neg_a=*/false, a, /*neg_b=*/false, bf,
            /*c_mod=*/(short)0, acc, /*reuse_a=*/false, /*reuse_b=*/false);
    }

    // C/D layout: lane&15 = N column, lane>>4 selects M rows 0-7 / 8-15.
    const int n     = nt * 16 + (lane & 15);
    const float bias = db[n];
    const int mbase = m0 + (lane >> 4) * 8;
#pragma unroll
    for (int v = 0; v < 8; ++v) {
        const int m  = mbase + v;
        const int b  = m >> 14;          // / HWn
        const int hw = m & (HWn - 1);
        out[((size_t)b * COUT + n) * HWn + hw] = acc[v] + bias;
    }
}

// ---------------------------------------------------------------------------
extern "C" void kernel_launch(void* const* d_in, const int* in_sizes, int n_in,
                              void* d_out, int out_size, void* d_ws, size_t ws_size,
                              hipStream_t stream) {
    const float* x    = (const float*)d_in[0];   // [8,64,128,128]
    const float* offw = (const float*)d_in[1];   // [1152,1,3,3]
    const float* offb = (const float*)d_in[2];   // [1152]
    const float* dw   = (const float*)d_in[3];   // [64,64,3,3]
    const float* db   = (const float*)d_in[4];   // [64]
    float* out        = (float*)d_out;           // [8,64,128,128]

    _Float16* Asamp = (_Float16*)d_ws;                      // M*CK f16
    _Float16* Bpack = Asamp + (size_t)Mn * CK;              // 36864 f16

    // Stage B0: pack weights (144 blocks).
    pack_w_kernel<<<(4 * KSTEPS * 512 + 255) / 256, 256, 0, stream>>>(dw, Bpack);

    // Stage A: one wave per pixel, 8 pixels per block -> 16384 blocks.
    deform_sample_kernel<<<Mn / 8, 256, 0, stream>>>(x, offw, offb, Asamp);

    // Stage B: 2 m-tiles x 4 n-tiles per block -> 4096 blocks.
    wmma_gemm_kernel<<<Mn / 32, 256, 0, stream>>>(Asamp, Bpack, db, out);
}
